// DIT_49881750175829
// MI455X (gfx1250) — matrix operand order repaired and checked by reading, not compiled, not run on hardware
//
#include <hip/hip_runtime.h>

#define Bq    2
#define Sq    2048
#define Nq    1024
#define DIMq  768
#define Hq    12
#define HDq   64
#define CONDq 128
#define NEGF  (-1e30f)

typedef __bf16 bf16;
typedef __attribute__((ext_vector_type(16))) __bf16 v16bf;
typedef __attribute__((ext_vector_type(8)))  float  v8f;

#if __has_builtin(__builtin_amdgcn_global_load_async_to_lds_b128)
#define HAVE_ASYNC_LDS 1
typedef __attribute__((ext_vector_type(4))) int i32x4;
typedef __attribute__((address_space(1))) i32x4 GI4;  // global-AS int4
typedef __attribute__((address_space(3))) i32x4 LI4;  // LDS-AS int4
#else
#define HAVE_ASYNC_LDS 0
#endif

static __device__ inline v8f wmma_bf16(v16bf a, v16bf b, v8f c) {
  // D = A(16x32) * B(32x16) + C, f32 accumulate
  return __builtin_amdgcn_wmma_f32_16x16x32_bf16(false, a, false, b, (short)0, c, false, false);
}

// A-fragment from a row-major 16x32 bf16 tile (4 uint4 per row).
// Layout used here (B matches, not the ISA table order): lane 0-15 = row M=lane; lane[4] selects K 0..15 vs 16..31 (contiguous).
// Those 8 dwords are one contiguous 32B run -> two b128 loads.
static __device__ inline v16bf load_A_frag(const uint4* base4, int lane) {
  const int row = lane & 15, kh = lane >> 4;
  union { uint4 q[2]; v16bf v; } t;
  t.q[0] = base4[row * 4 + 2 * kh];       // words kh*8 .. kh*8+3
  t.q[1] = base4[row * 4 + 2 * kh + 1];   // words kh*8+4 .. kh*8+7
  return t.v;
}

// B-fragment: W is (N,K) row-major bf16, tile at (n0,k0). B[k][n] = W[n0+n][k0+k].
// Lane n holds column n; lane[4] selects K 0..15 / 16..31 -> 32 contiguous bytes per lane.
static __device__ inline v16bf load_B_rowmajor(const bf16* W, size_t ldw, int n0, int k0, int lane) {
  const int col = lane & 15, kh = lane >> 4;
  const uint4* p = reinterpret_cast<const uint4*>(W + (size_t)(n0 + col) * ldw + k0 + kh * 16);
  union { uint4 u[2]; v16bf v; } t;
  t.u[0] = p[0]; t.u[1] = p[1];
  return t.v;
}

static __device__ inline bool attn_mask(int q, int k) {
  const int flq = (q >= Nq), flk = (k >= Nq);
  const int bq = flq ? ((q - Nq) >> 4) : (q >> 4);
  const int bk = flk ? ((k - Nq) >> 4) : (k >> 4);
  const bool same    = (bq == bk) && (flq == flk);
  const bool allowed = (bq > 0) && (bk < bq) && (flk == 1);
  return same || allowed;
}

// ---------------- small prep kernels ----------------

__global__ __launch_bounds__(256) void mods_kernel(const float* __restrict__ c,
                                                   const float* __restrict__ aw,
                                                   const float* __restrict__ ab,
                                                   float* __restrict__ mods) {
  const int j = blockIdx.x * 256 + threadIdx.x;
  if (j >= Bq * 6 * DIMq) return;
  const int b = j / (6 * DIMq), n = j - b * (6 * DIMq);
  const float* cp = c + b * CONDq;
  const float* wp = aw + (size_t)n * CONDq;
  float s = ab[n];
  for (int k = 0; k < CONDq; ++k) s = fmaf(cp[k], wp[k], s);
  mods[j] = s;
}

__global__ __launch_bounds__(256) void f2bf_kernel(const float* __restrict__ src,
                                                   bf16* __restrict__ dst, int n) {
  const int i = blockIdx.x * 256 + threadIdx.x;
  if (i < n) dst[i] = (bf16)src[i];
}

// LayerNorm + adaLN modulation -> bf16 activations
__global__ __launch_bounds__(256) void ln_mod_kernel(const float* __restrict__ x,
                                                     const float* __restrict__ w,
                                                     const float* __restrict__ mods,
                                                     int shiftIdx, int scaleIdx,
                                                     bf16* __restrict__ out) {
  __shared__ float red[256];
  const int row = blockIdx.x, tid = threadIdx.x;
  const int b = row >> 11; // row / S
  const float* xr = x + (size_t)row * DIMq;
  float s = 0.f;
  for (int i = tid; i < DIMq; i += 256) s += xr[i];
  red[tid] = s; __syncthreads();
  for (int off = 128; off > 0; off >>= 1) { if (tid < off) red[tid] += red[tid + off]; __syncthreads(); }
  const float mu = red[0] / (float)DIMq;
  __syncthreads();
  float vs = 0.f;
  for (int i = tid; i < DIMq; i += 256) { const float d = xr[i] - mu; vs += d * d; }
  red[tid] = vs; __syncthreads();
  for (int off = 128; off > 0; off >>= 1) { if (tid < off) red[tid] += red[tid + off]; __syncthreads(); }
  const float rstd = rsqrtf(red[0] / (float)DIMq + 1e-5f);
  const float* mb = mods + b * 6 * DIMq;
  for (int i = tid; i < DIMq; i += 256) {
    const float hv = (xr[i] - mu) * rstd * w[i] * (1.f + mb[scaleIdx * DIMq + i]) + mb[shiftIdx * DIMq + i];
    out[(size_t)row * DIMq + i] = (bf16)hv;
  }
}

// RoPE on raw qkv (applied to q, k AND v, matching the reference).
// q,k -> [b,s,h,d] (row = contiguous over d, feeds A/B fragments of Q@K^T).
// v   -> TRANSPOSED [b,h,d,s] so the P@V B-fragment (contiguous over keys) is 2x b128.
__global__ __launch_bounds__(256) void rope_kernel(const bf16* __restrict__ qkv,
                                                   const float* __restrict__ cosb,
                                                   const float* __restrict__ sinb,
                                                   bf16* __restrict__ qb,
                                                   bf16* __restrict__ kb,
                                                   bf16* __restrict__ vtb) {
  const int idx = blockIdx.x * 256 + threadIdx.x;
  const int total = (Bq * Sq) * (3 * DIMq);
  if (idx >= total) return;
  const int m = idx / (3 * DIMq), n = idx - m * (3 * DIMq);
  const int t = n / DIMq, rem = n - t * DIMq;
  const int h = rem >> 6, d = rem & 63;
  const int s = m & (Sq - 1);
  const int b = m >> 11;
  const float v = (float)qkv[idx];
  const int dp = (d < 32) ? d + 32 : d - 32;
  const float other = (float)qkv[(size_t)m * (3 * DIMq) + t * DIMq + h * 64 + dp];
  const float rot = (d < 32) ? -other : other;
  const float o = v * cosb[s * HDq + d] + rot * sinb[s * HDq + d];
  const bf16 ov = (bf16)o;
  if (t == 2) {
    vtb[(((size_t)b * Hq + h) * HDq + d) * Sq + s] = ov;               // V^T
  } else {
    const size_t oi = ((size_t)m * Hq + h) * HDq + d;                  // [b,s,h,d]
    if (t == 0) qb[oi] = ov; else kb[oi] = ov;
  }
}

// ---------------- WMMA GEMM: out(M,N) = A(M,K) @ W(N,K)^T + epilogue ----------------
// Block = 8 waves covering 64 rows x 128 cols. Each wave owns a 64x16 column strip:
// the B fragment loaded per K-step is reused by 4 WMMAs (4x arithmetic intensity vs 16x16).
// A tile (64x32 bf16, 4KB) staged in LDS via ASYNCcnt-tracked async loads when available.

template <int EP>
__global__ __launch_bounds__(256) void gemm_wmma(const bf16* __restrict__ A,
                                                 const bf16* __restrict__ W,
                                                 int M, int Nn, int K,
                                                 float* __restrict__ outF,
                                                 bf16* __restrict__ outB,
                                                 const float* __restrict__ mods,
                                                 const float* __restrict__ bias,
                                                 const float* __restrict__ skip) {
  __shared__ __align__(16) uint4 sA[64 * 4]; // 64 rows x 32 bf16
  const int tid = threadIdx.x, lane = tid & 31, wave = tid >> 5;
  const int m0 = blockIdx.y * 64;
  const int n0 = (blockIdx.x * 8 + wave) * 16;
  const int srow = tid >> 2, squad = tid & 3; // staging: 1 uint4 per thread
  v8f acc[4];
  {
    v8f z = {};
#pragma unroll
    for (int mt = 0; mt < 4; ++mt) acc[mt] = z;
  }
  for (int k0 = 0; k0 < K; k0 += 32) {
    const uint4* gp = reinterpret_cast<const uint4*>(A + (size_t)(m0 + srow) * K + k0) + squad;
#if HAVE_ASYNC_LDS
    __builtin_amdgcn_global_load_async_to_lds_b128((GI4*)gp, (LI4*)&sA[tid], 0, 0);
    asm volatile("s_wait_asynccnt 0x0" ::: "memory");
#else
    sA[tid] = gp[0];
#endif
    __syncthreads();
    const v16bf bv = load_B_rowmajor(W, (size_t)K, n0, k0, lane);
#pragma unroll
    for (int mt = 0; mt < 4; ++mt) {
      const v16bf av = load_A_frag(&sA[mt * 16 * 4], lane);
      acc[mt] = wmma_bf16(av, bv, acc[mt]);
    }
    __syncthreads();
  }
  const int cn = lane & 15, kh = lane >> 4;
#pragma unroll
  for (int mt = 0; mt < 4; ++mt) {
#pragma unroll
    for (int r = 0; r < 8; ++r) {
      const int m = m0 + mt * 16 + r + 8 * kh;
      const int n = n0 + cn;
      const size_t o = (size_t)m * Nn + n;
      float v = acc[mt][r];
      if (EP == 0) {                      // raw qkv -> bf16
        outB[o] = (bf16)v;
      } else if (EP == 1) {               // x = gate_msa * (a@Wo^T) + x_skip
        const int b = m >> 11;
        const float gate = mods[b * 6 * DIMq + 2 * DIMq + n];
        outF[o] = gate * v + skip[o];
      } else if (EP == 2) {               // gelu(h2@W1^T + b1) -> bf16
        v += bias[n];
        const float t = 0.7978845608028654f * (v + 0.044715f * v * v * v);
        outB[o] = (bf16)(0.5f * v * (1.0f + tanhf(t)));
      } else {                            // out = x + gate_mlp * (m1@W2^T + b2)
        const int b = m >> 11;
        v += bias[n];
        const float gate = mods[b * 6 * DIMq + 5 * DIMq + n];
        outF[o] = skip[o] + gate * v;
      }
    }
  }
}

// ---------------- flash attention: one wave = one 16-query tile of one (b,h) ----------------

__global__ __launch_bounds__(256) void attn_kernel(const bf16* __restrict__ qb,
                                                   const bf16* __restrict__ kb,
                                                   const bf16* __restrict__ vtb,
                                                   bf16* __restrict__ ab) {
  __shared__ __align__(16) bf16 pl[8][16 * 32]; // per-wave P tile (16 rows x 32 keys)
  const int tid = threadIdx.x, lane = tid & 31, wave = tid >> 5;
  const int tile = blockIdx.x * 8 + wave;       // b*H*128 + h*128 + qt
  const int qt = tile & 127;
  const int h  = (tile >> 7) % Hq;
  const int b  = tile / (128 * Hq);
  const int q0 = qt * 16;
  const int row = lane & 15, kh = lane >> 4, cn = row;

  // Q fragments: 16x64 split into two K=32 A-fragments; each = 2x b128 (row = 8 uint4)
  v16bf qa0, qa1;
  {
    const uint4* qp = reinterpret_cast<const uint4*>(
        qb + (((size_t)(b * Sq + q0 + row)) * Hq + h) * HDq);
    union { uint4 q[2]; v16bf v; } t0, t1;
    t0.q[0] = qp[2 * kh];     t0.q[1] = qp[2 * kh + 1];   // d = 0..31
    t1.q[0] = qp[4 + 2 * kh]; t1.q[1] = qp[5 + 2 * kh];   // d = 32..63
    qa0 = t0.v; qa1 = t1.v;
  }

  float m_i[8], l_i[8];
  v8f accO[4];
  {
    v8f z = {};
#pragma unroll
    for (int nt = 0; nt < 4; ++nt) accO[nt] = z;
#pragma unroll
    for (int r = 0; r < 8; ++r) { m_i[r] = NEGF; l_i[r] = 0.f; }
  }

  for (int k0 = 0; k0 < Sq; k0 += 32) {
    v8f sc[2];
#pragma unroll
    for (int t = 0; t < 2; ++t) { // two 16-key score tiles
      v8f s = {};
#pragma unroll
      for (int c = 0; c < 2; ++c) { // HD = 64 -> two K=32 steps
        const uint4* kp = reinterpret_cast<const uint4*>(
            kb + (((size_t)(b * Sq + k0 + t * 16 + cn)) * Hq + h) * HDq + c * 32 + kh * 16);
        union { uint4 u[2]; v16bf v; } tb;
        tb.u[0] = kp[0]; tb.u[1] = kp[1];
        s = wmma_bf16(c ? qa1 : qa0, tb.v, s);
      }
      sc[t] = s;
    }
    // online softmax over the 32 new keys (row stats via 16-lane xor shuffles within each half)
    float alpha[8];
#pragma unroll
    for (int r = 0; r < 8; ++r) {
      const int qi = q0 + r + 8 * kh;
      const bool ok0 = attn_mask(qi, k0 + cn);
      const bool ok1 = attn_mask(qi, k0 + 16 + cn);
      float v0 = ok0 ? sc[0][r] * 0.125f : NEGF; // 1/sqrt(64)
      float v1 = ok1 ? sc[1][r] * 0.125f : NEGF;
      float mx = fmaxf(v0, v1);
#pragma unroll
      for (int off = 1; off < 16; off <<= 1) mx = fmaxf(mx, __shfl_xor(mx, off, 32));
      const float mnew = fmaxf(m_i[r], mx);
      const float a = __expf(m_i[r] - mnew);
      const float p0 = ok0 ? __expf(v0 - mnew) : 0.f;
      const float p1 = ok1 ? __expf(v1 - mnew) : 0.f;
      float sm = p0 + p1;
#pragma unroll
      for (int off = 1; off < 16; off <<= 1) sm += __shfl_xor(sm, off, 32);
      l_i[r] = l_i[r] * a + sm;
      m_i[r] = mnew;
      alpha[r] = a;
      const int lr = r + 8 * kh;
      pl[wave][lr * 32 + cn]      = (bf16)p0; // C-layout -> row-major LDS
      pl[wave][lr * 32 + 16 + cn] = (bf16)p1;
    }
#pragma unroll
    for (int nt = 0; nt < 4; ++nt)
#pragma unroll
      for (int r = 0; r < 8; ++r) accO[nt][r] *= alpha[r];

    // P (16x32) as A-fragment from LDS (2x ds b128); V^T B-fragments are 2x global b128.
    const v16bf pv = load_A_frag(reinterpret_cast<const uint4*>(&pl[wave][0]), lane);
#pragma unroll
    for (int nt = 0; nt < 4; ++nt) {
      const uint4* vp = reinterpret_cast<const uint4*>(
          vtb + (((size_t)b * Hq + h) * HDq + nt * 16 + cn) * Sq + k0 + kh * 16);
      union { uint4 u[2]; v16bf v; } tb;
      tb.u[0] = vp[0]; tb.u[1] = vp[1];
      accO[nt] = wmma_bf16(pv, tb.v, accO[nt]);
    }
  }

#pragma unroll
  for (int r = 0; r < 8; ++r) {
    const float inv = 1.f / l_i[r];
    const int s = q0 + r + 8 * kh;
#pragma unroll
    for (int nt = 0; nt < 4; ++nt) {
      const int d = nt * 16 + cn;
      ab[(size_t)(b * Sq + s) * DIMq + h * HDq + d] = (bf16)(accO[nt][r] * inv);
    }
  }
}

// ---------------- host orchestration ----------------

extern "C" void kernel_launch(void* const* d_in, const int* in_sizes, int n_in,
                              void* d_out, int out_size, void* d_ws, size_t ws_size,
                              hipStream_t stream) {
  (void)in_sizes; (void)n_in; (void)out_size; (void)ws_size;
  const float* x          = (const float*)d_in[0];
  const float* c          = (const float*)d_in[1];
  const float* cosb       = (const float*)d_in[2];
  const float* sinb       = (const float*)d_in[3];
  /* d_in[4] = mask: computed analytically in-kernel */
  const float* norm1_w    = (const float*)d_in[5];
  const float* qkv_w      = (const float*)d_in[6];
  const float* attn_out_w = (const float*)d_in[7];
  const float* norm2_w    = (const float*)d_in[8];
  const float* mlp_w1     = (const float*)d_in[9];
  const float* mlp_b1     = (const float*)d_in[10];
  const float* mlp_w2     = (const float*)d_in[11];
  const float* mlp_b2     = (const float*)d_in[12];
  const float* adaLN_w    = (const float*)d_in[13];
  const float* adaLN_b    = (const float*)d_in[14];

  char* ws = (char*)d_ws;
  size_t off = 0;
  auto carve = [&](size_t bytes) -> void* {
    void* p = ws + off;
    off += (bytes + 255) & ~(size_t)255;
    return p;
  };
  const size_t MR = (size_t)Bq * Sq; // 4096 rows
  float* modsb  = (float*)carve(Bq * 6 * DIMq * sizeof(float));
  bf16*  wq     = (bf16*)carve((size_t)3 * DIMq * DIMq * 2);
  bf16*  wo     = (bf16*)carve((size_t)DIMq * DIMq * 2);
  bf16*  w1     = (bf16*)carve((size_t)4 * DIMq * DIMq * 2);
  bf16*  w2     = (bf16*)carve((size_t)DIMq * 4 * DIMq * 2);
  bf16*  hbuf   = (bf16*)carve(MR * DIMq * 2);
  bf16*  qkvraw = (bf16*)carve(MR * 3 * DIMq * 2);
  bf16*  qbuf   = (bf16*)carve(MR * DIMq * 2);
  bf16*  kbuf   = (bf16*)carve(MR * DIMq * 2);
  bf16*  vtbuf  = (bf16*)carve(MR * DIMq * 2);   // V transposed [b,h,d,s]
  bf16*  abuf   = (bf16*)carve(MR * DIMq * 2);
  float* xws    = (float*)carve(MR * DIMq * sizeof(float));
  bf16*  h2     = (bf16*)carve(MR * DIMq * 2);
  bf16*  m1     = (bf16*)carve(MR * 4 * DIMq * 2);

  const int M = (int)MR;

  mods_kernel<<<(Bq * 6 * DIMq + 255) / 256, 256, 0, stream>>>(c, adaLN_w, adaLN_b, modsb);
  f2bf_kernel<<<(3 * DIMq * DIMq + 255) / 256, 256, 0, stream>>>(qkv_w, wq, 3 * DIMq * DIMq);
  f2bf_kernel<<<(DIMq * DIMq + 255) / 256, 256, 0, stream>>>(attn_out_w, wo, DIMq * DIMq);
  f2bf_kernel<<<(4 * DIMq * DIMq + 255) / 256, 256, 0, stream>>>(mlp_w1, w1, 4 * DIMq * DIMq);
  f2bf_kernel<<<(DIMq * 4 * DIMq + 255) / 256, 256, 0, stream>>>(mlp_w2, w2, DIMq * 4 * DIMq);

  ln_mod_kernel<<<M, 256, 0, stream>>>(x, norm1_w, modsb, 0, 1, hbuf);

  gemm_wmma<0><<<dim3(3 * DIMq / 128, M / 64), 256, 0, stream>>>(
      hbuf, wq, M, 3 * DIMq, DIMq, nullptr, qkvraw, nullptr, nullptr, nullptr);

  rope_kernel<<<(M * 3 * DIMq + 255) / 256, 256, 0, stream>>>(qkvraw, cosb, sinb, qbuf, kbuf, vtbuf);

  attn_kernel<<<(Bq * Hq * (Sq / 16)) / 8, 256, 0, stream>>>(qbuf, kbuf, vtbuf, abuf);

  gemm_wmma<1><<<dim3(DIMq / 128, M / 64), 256, 0, stream>>>(
      abuf, wo, M, DIMq, DIMq, xws, nullptr, modsb, nullptr, x);

  ln_mod_kernel<<<M, 256, 0, stream>>>(xws, norm2_w, modsb, 3, 4, h2);

  gemm_wmma<2><<<dim3(4 * DIMq / 128, M / 64), 256, 0, stream>>>(
      h2, w1, M, 4 * DIMq, DIMq, nullptr, m1, nullptr, mlp_b1, nullptr);

  gemm_wmma<3><<<dim3(DIMq / 128, M / 64), 256, 0, stream>>>(
      m1, w2, M, DIMq, 4 * DIMq, (float*)d_out, nullptr, modsb, mlp_b2, xws);
}
